// STGNNModel_24163486007581
// MI455X (gfx1250) — compile-verified, matrix operand order checked by reading, not run on hardware
//
#include <hip/hip_runtime.h>
#include <hip/hip_bf16.h>

typedef __attribute__((ext_vector_type(16))) _Float16 v16h;
typedef __attribute__((ext_vector_type(8)))  float    v8f;

#define Bdim 4
#define Tdim 24
#define Ndim 10000
#define Edim 160000
#define SH   16
#define TH   32
#define BT   (Bdim * Tdim)        // 96
#define BN   (Bdim * Ndim)        // 40000
#define G3   (3 * TH)             // 96 gates (r,z,n)

#define ROWS_PER_BLOCK 128        // 8 waves * 16 rows
#define HSTRIDE 34                // even padded stride: 8B-aligned K-pairs, banks 34*i mod 64 distinct

// Branch-free fast activations: v_exp_f32 + v_rcp_f32, no IEEE div, no execz branches.
__device__ __forceinline__ float fast_sigmoid(float x) {
    return __builtin_amdgcn_rcpf(1.0f + __expf(-x));
}
__device__ __forceinline__ float fast_tanh(float x) {
    // tanh(x) = 1 - 2/(exp(2x)+1); exp overflow -> inf -> rcp=0 -> +1, underflow -> -1.
    return 1.0f - 2.0f * __builtin_amdgcn_rcpf(__expf(2.0f * x) + 1.0f);
}

// ---------- xT[n*BT + bt] = x[bt*N + n] ----------
__global__ void transpose_kernel(const float* __restrict__ x, float* __restrict__ xT) {
    int i = blockIdx.x * 256 + threadIdx.x;          // over N*BT, (n major, bt minor)
    if (i < Ndim * BT) {
        int n = i / BT, bt = i % BT;
        xT[i] = x[(long long)bt * Ndim + n];
    }
}

// ---------- deg[dst] += 1 ----------
__global__ void deg_kernel(const int* __restrict__ ei, float* __restrict__ deg) {
    int e = blockIdx.x * 256 + threadIdx.x;
    if (e < Edim) atomicAdd(&deg[ei[Edim + e]], 1.0f);   // ei[1][e] = dst
}

// ---------- dinv = rsqrt(deg + 1) (self loop) ----------
__global__ void dinv_kernel(const float* __restrict__ deg, float* __restrict__ dinv) {
    int n = blockIdx.x * 256 + threadIdx.x;
    if (n < Ndim) dinv[n] = rsqrtf(deg[n] + 1.0f);
}

// ---------- scalar edge aggregation: aggs[dst][bt] += x[bt,src]*norm ----------
__global__ void edge_agg_kernel(const int* __restrict__ ei, const float* __restrict__ xT,
                                const float* __restrict__ dinv, float* __restrict__ aggs) {
    unsigned i = blockIdx.x * 256u + threadIdx.x;    // over E*BT, (e major, bt minor)
    if (i < (unsigned)Edim * BT) {
        int e  = i / BT, bt = i % BT;
        int src = ei[e], dst = ei[Edim + e];
        float nrm = dinv[src] * dinv[dst];
        atomicAdd(&aggs[(long long)dst * BT + bt], xT[(long long)src * BT + bt] * nrm);
    }
}

// ---------- fold GCN weights + biases into GRU projection ----------
// u[g] = (W_ih @ w_gcn)[g]
// p[g] = (W_ih @ b_gcn)[g] + b_ih[g] + (g<64 ? b_hh[g] : 0)   (r,z absorb b_hh; n keeps it)
// q[j] = b_hh[64+j]                                           (n-gate hidden bias, scaled by r)
__global__ void uv_kernel(const float* __restrict__ W_ih, const float* __restrict__ w_gcn,
                          const float* __restrict__ b_gcn, const float* __restrict__ b_ih,
                          const float* __restrict__ b_hh,
                          float* __restrict__ u, float* __restrict__ p, float* __restrict__ q) {
    int g = threadIdx.x;
    if (g < G3) {
        float su = 0.f, sv = 0.f;
        #pragma unroll
        for (int f = 0; f < SH; ++f) {
            float w = W_ih[g * SH + f];
            su += w * w_gcn[f];
            sv += w * b_gcn[f];
        }
        u[g] = su;
        p[g] = sv + b_ih[g] + ((g < 2 * TH) ? b_hh[g] : 0.0f);
        if (g < TH) q[g] = b_hh[2 * TH + g];
    }
}

// ---------- s[t*BN + b*N + n] = aggs[n][b*T+t] + xT[n][b*T+t]*dinv[n]^2 ----------
__global__ void s_kernel(const float* __restrict__ aggs, const float* __restrict__ xT,
                         const float* __restrict__ dinv, float* __restrict__ s) {
    int i = blockIdx.x * 256 + threadIdx.x;          // layout (t, b, n): i = t*B*N + b*N + n
    if (i < Tdim * Bdim * Ndim) {
        int n  = i % Ndim;
        int tb = i / Ndim;
        int b  = tb % Bdim;
        int t  = tb / Bdim;
        int bt = b * Tdim + t;
        float dv = dinv[n];
        s[i] = aggs[(long long)n * BT + bt] + xT[(long long)n * BT + bt] * dv * dv;
    }
}

// ---------- fused WMMA GRU scan + FC head ----------
// One wave owns 16 batch rows; per step: gh = h(f16 16x32) x W_hh^T(f16 32x96) via
// 6x v_wmma_f32_16x16x32_f16, then branch-free gate math off the accumulator layout.
__global__ __launch_bounds__(256) void gru_kernel(
    const float* __restrict__ s, const float* __restrict__ whh,
    const float* __restrict__ u, const float* __restrict__ p,
    const float* __restrict__ q, const float* __restrict__ wfc,
    const float* __restrict__ bfc, float* __restrict__ out) {

    __shared__ float h_lds[ROWS_PER_BLOCK * HSTRIDE];
    __shared__ float whh_lds[G3 * TH];
    __shared__ float u_lds[G3], p_lds[G3], q_lds[TH], wfc_lds[TH];
    __shared__ float s_lds[ROWS_PER_BLOCK];

    const int tid = threadIdx.x;
    for (int i = tid; i < G3 * TH; i += 256) whh_lds[i] = whh[i];
    if (tid < G3) { u_lds[tid] = u[tid]; p_lds[tid] = p[tid]; }
    if (tid < TH) { q_lds[tid] = q[tid]; wfc_lds[tid] = wfc[tid]; }
    for (int i = tid; i < ROWS_PER_BLOCK * HSTRIDE; i += 256) h_lds[i] = 0.0f;
    __syncthreads();

    const int wave     = tid >> 5;
    const int lane     = tid & 31;
    const int hi       = lane >> 4;          // lane half (ISA A/B/C fragment split)
    const int c        = lane & 15;          // column within 16-wide tile
    const int waveBase = wave * 16;          // this wave's first local row
    const long long blockBase = (long long)blockIdx.x * ROWS_PER_BLOCK;

    // B fragments: B[k][n] = W_hh[ n_global ][ k ]  (W_hh^T), 16x16x32 f16 layout.
    // K-chunk map (ISA 16-bit A/B 16x32): vgpr v, half hi -> k0 = (v<4 ? 2v : 16+2(v-4)) + 8*hi
    v16h bfrag[6];
    #pragma unroll
    for (int jt = 0; jt < 6; ++jt) {
        const int col = jt * 16 + c;                      // gate index 0..95
        #pragma unroll
        for (int v = 0; v < 8; ++v) {
            const int k0 = ((v < 4) ? (v * 2) : (16 + (v - 4) * 2)) + hi * 8;
            bfrag[jt][2 * v]     = (_Float16)whh_lds[col * TH + k0];
            bfrag[jt][2 * v + 1] = (_Float16)whh_lds[col * TH + k0 + 1];
        }
    }

    const int arow = waveBase + c;           // this lane's A-matrix row (M = lane&15)

    for (int t = 0; t < Tdim; ++t) {
        // stage this wave's 16 s-values (coalesced per-wave)
        if (lane < 16) {
            long long gr = blockBase + waveBase + lane;
            s_lds[waveBase + lane] = (gr < BN) ? s[(long long)t * BN + gr] : 0.0f;
        }

        // A fragment: h row 'arow', f32 -> f16; 8B-aligned K-pairs -> ds_load_b64
        v16h a;
        #pragma unroll
        for (int v = 0; v < 8; ++v) {
            const int k0 = ((v < 4) ? (v * 2) : (16 + (v - 4) * 2)) + hi * 8;
            const float2 hv = *reinterpret_cast<const float2*>(&h_lds[arow * HSTRIDE + k0]);
            a[2 * v]     = (_Float16)hv.x;
            a[2 * v + 1] = (_Float16)hv.y;
        }

        // gh tiles: 6 x (16 rows x 16 gate cols), f32 accumulate
        v8f d[6];
        #pragma unroll
        for (int jt = 0; jt < 6; ++jt) {
            v8f cz = {};
            d[jt] = __builtin_amdgcn_wmma_f32_16x16x32_f16(
                false, a, false, bfrag[jt], (short)0, cz, false, false);
        }

        // Gate update. D layout: vgpr v holds row (v + 8*hi), lane holds column c of
        // each tile -> this lane owns gate columns {c, c+16} of r/z/n for 8 rows.
        #pragma unroll
        for (int v = 0; v < 8; ++v) {
            const int mrow = waveBase + v + 8 * hi;
            const float sv = s_lds[mrow];
            // column g = c   : tiles 0 (r), 2 (z), 4 (n)
            {
                const int g = c;
                float r  = fast_sigmoid(fmaf(sv, u_lds[g],      p_lds[g])      + d[0][v]);
                float z  = fast_sigmoid(fmaf(sv, u_lds[32 + g], p_lds[32 + g]) + d[2][v]);
                float nn = fast_tanh(fmaf(sv, u_lds[64 + g], p_lds[64 + g]) + r * (d[4][v] + q_lds[g]));
                float ho = h_lds[mrow * HSTRIDE + g];
                h_lds[mrow * HSTRIDE + g] = fmaf(z, ho - nn, nn);   // (1-z)*n + z*h
            }
            // column g = c+16: tiles 1 (r), 3 (z), 5 (n)
            {
                const int g = c + 16;
                float r  = fast_sigmoid(fmaf(sv, u_lds[g],      p_lds[g])      + d[1][v]);
                float z  = fast_sigmoid(fmaf(sv, u_lds[32 + g], p_lds[32 + g]) + d[3][v]);
                float nn = fast_tanh(fmaf(sv, u_lds[64 + g], p_lds[64 + g]) + r * (d[5][v] + q_lds[g]));
                float ho = h_lds[mrow * HSTRIDE + g];
                h_lds[mrow * HSTRIDE + g] = fmaf(z, ho - nn, nn);
            }
        }
        // no block barrier needed: each wave touches only its own h/s rows (LDS is
        // in-order per wave, wave32 lockstep)
    }

    __syncthreads();
    // FC head: out[row] = h_last . W_fc + b_fc   (rows 0..127 read cross-wave)
    if (tid < ROWS_PER_BLOCK) {
        long long gr = blockBase + tid;
        if (gr < BN) {
            float acc = bfc[0];
            #pragma unroll
            for (int k = 0; k < TH; ++k) acc += h_lds[tid * HSTRIDE + k] * wfc_lds[k];
            out[gr] = acc;
        }
    }
}

extern "C" void kernel_launch(void* const* d_in, const int* in_sizes, int n_in,
                              void* d_out, int out_size, void* d_ws, size_t ws_size,
                              hipStream_t stream) {
    const float* x     = (const float*)d_in[0];
    const int*   ei    = (const int*)  d_in[1];
    const float* w_gcn = (const float*)d_in[2];
    const float* b_gcn = (const float*)d_in[3];
    const float* W_ih  = (const float*)d_in[4];
    const float* W_hh  = (const float*)d_in[5];
    const float* b_ih  = (const float*)d_in[6];
    const float* b_hh  = (const float*)d_in[7];
    const float* W_fc  = (const float*)d_in[8];
    const float* b_fc  = (const float*)d_in[9];
    float* out = (float*)d_out;

    float* ws   = (float*)d_ws;
    float* deg  = ws;                  ws += Ndim;
    float* dinv = ws;                  ws += Ndim;
    float* xT   = ws;                  ws += (size_t)Ndim * BT;
    float* aggs = ws;                  ws += (size_t)Ndim * BT;
    float* u    = ws;                  ws += G3;
    float* p    = ws;                  ws += G3;
    float* q    = ws;                  ws += TH;
    float* s    = ws;                  ws += (size_t)Tdim * BN;

    hipMemsetAsync(deg,  0, (size_t)Ndim * sizeof(float), stream);
    hipMemsetAsync(aggs, 0, (size_t)Ndim * BT * sizeof(float), stream);

    transpose_kernel<<<(Ndim * BT + 255) / 256, 256, 0, stream>>>(x, xT);
    deg_kernel      <<<(Edim + 255) / 256,      256, 0, stream>>>(ei, deg);
    dinv_kernel     <<<(Ndim + 255) / 256,      256, 0, stream>>>(deg, dinv);
    edge_agg_kernel <<<((unsigned)Edim * BT + 255) / 256, 256, 0, stream>>>(ei, xT, dinv, aggs);
    uv_kernel       <<<1, 128, 0, stream>>>(W_ih, w_gcn, b_gcn, b_ih, b_hh, u, p, q);
    s_kernel        <<<(Tdim * Bdim * Ndim + 255) / 256, 256, 0, stream>>>(aggs, xT, dinv, s);
    gru_kernel      <<<(BN + ROWS_PER_BLOCK - 1) / ROWS_PER_BLOCK, 256, 0, stream>>>(
        s, W_hh, u, p, q, W_fc, b_fc, out);
}
